// StyleBlock_34540126994989
// MI455X (gfx1250) — matrix operand order repaired and checked by reading, not compiled, not run on hardware
//
#include <hip/hip_runtime.h>
#include <hip/hip_bf16.h>
#include <math.h>

#define NC     256
#define HH     128
#define WW     128
#define PP     (HH*WW)        // 16384
#define SD     512
#define KCK    (NC*9)         // 2304
#define TILE_P 32
#define KC     64             // K chunk (one conv tap, 64 channels)
#define NKC    (KCK/KC)       // 36
#define BATCH  8
#define APITCH 72             // padded LDS row pitch in halves (144B)
#define BPITCH 72

typedef __attribute__((ext_vector_type(16))) __bf16        v16bf;
typedef __attribute__((ext_vector_type(8)))  float          v8f;
typedef __attribute__((ext_vector_type(16))) unsigned short v16u;
typedef v16u __attribute__((aligned(16)))                   v16u_a16;

static __device__ __forceinline__ unsigned short f2bf(float f) {
  unsigned int u = __float_as_uint(f);
  u += 0x7FFFu + ((u >> 16) & 1u);          // round-to-nearest-even
  return (unsigned short)(u >> 16);
}

// ---------------------------------------------------------------------------
// Style kernel: s = 1 + PixelNorm(EqualLinear(latent)); d = rsqrt(sum (s*w)^2)
// grid: 16 blocks (stage*8 + batch), 256 threads (one per channel)
// ---------------------------------------------------------------------------
__global__ void style_kernel(const float* __restrict__ latent,
                             const float* __restrict__ el0_w, const float* __restrict__ el0_b,
                             const float* __restrict__ el1_w, const float* __restrict__ el1_b,
                             const float* __restrict__ w0, const float* __restrict__ w1,
                             float* __restrict__ sArr, float* __restrict__ dArr) {
  const int stage = blockIdx.x >> 3;
  const int b     = blockIdx.x & 7;
  const int t     = threadIdx.x;
  const float* elw = stage ? el1_w : el0_w;
  const float* elb = stage ? el1_b : el0_b;
  const float* wgt = stage ? w1    : w0;

  float e = 0.f;
  const float* lrow = latent + b * SD;
  const float* wrow = elw + t * SD;
  for (int i = 0; i < SD; ++i) e += lrow[i] * wrow[i];
  e = e * 0.044194173824159216f /* 1/sqrt(512) */ + elb[t];

  __shared__ float red[NC];
  __shared__ float shS[NC];
  red[t] = e * e;
  __syncthreads();
  for (int off = 128; off > 0; off >>= 1) {
    if (t < off) red[t] += red[t + off];
    __syncthreads();
  }
  const float mean = red[0] * (1.f / NC);
  const float s = 1.f + e * rsqrtf(mean + 1e-8f);
  shS[t] = s;
  sArr[(stage * BATCH + b) * NC + t] = s;
  __syncthreads();

  // demodulation for out-channel t
  const float g2 = 2.f / (float)KCK;
  float sum = 0.f;
  const float* wo = wgt + (size_t)t * NC * 9;
  for (int c = 0; c < NC; ++c) {
    float wsq = 0.f;
#pragma unroll
    for (int k = 0; k < 9; ++k) { float wv = wo[c * 9 + k]; wsq += wv * wv; }
    float sc = shS[c];
    sum += sc * sc * wsq * g2;
  }
  dArr[(stage * BATCH + b) * NC + t] = rsqrtf(sum + 1e-5f);
}

// ---------------------------------------------------------------------------
// Weight prep: bf16, TRANSPOSED oc-major layout wbT[oc][ck2], ck2 = k*256 + c
// (tap-major K order), scale sqrt(2/fan_in) folded in.
// ---------------------------------------------------------------------------
__global__ void wprep_kernel(const float* __restrict__ w0, const float* __restrict__ w1,
                             unsigned short* __restrict__ wb0, unsigned short* __restrict__ wb1) {
  const int stage = blockIdx.y;
  const float* wsrc = stage ? w1 : w0;
  unsigned short* wdst = stage ? wb1 : wb0;
  const float g = sqrtf(2.f / (float)KCK);
  for (int idx = blockIdx.x * blockDim.x + threadIdx.x; idx < NC * KCK;
       idx += gridDim.x * blockDim.x) {
    int oc  = idx / KCK;
    int ck2 = idx - oc * KCK;
    int k = ck2 >> 8;          // tap 0..8
    int c = ck2 & 255;         // input channel
    wdst[idx] = f2bf(wsrc[((size_t)oc * NC + c) * 9 + k] * g);
  }
}

// ---------------------------------------------------------------------------
// Modulate input with s0 and pack to bf16
// ---------------------------------------------------------------------------
__global__ void modulate_kernel(const float* __restrict__ x, const float* __restrict__ s0,
                                unsigned short* __restrict__ xm) {
  const long long total = (long long)BATCH * NC * PP;
  for (long long idx = (long long)blockIdx.x * blockDim.x + threadIdx.x; idx < total;
       idx += (long long)gridDim.x * blockDim.x) {
    int bc = (int)(idx >> 14);     // b*NC + c
    xm[idx] = f2bf(x[idx] * s0[bc]);
  }
}

// ---------------------------------------------------------------------------
// Modulated conv 3x3 as implicit GEMM (bf16 WMMA, f32 accum).
// WG: 256 threads (8 waves), tile = 32 spatial x 256 oc, K = 2304 in chunks
// of 64 (one conv tap each). Fused epilogue: demod, bias, leaky*sqrt2,
// cross-channel pixel-norm, then (stage0) *s1 + bf16 pack or (stage1) f32 out.
// ---------------------------------------------------------------------------
__global__ __launch_bounds__(256, 2)
void modconv_kernel(const unsigned short* __restrict__ xm,   // bf16 [B][NC][PP]
                    const unsigned short* __restrict__ wbT,  // bf16 [NC][KCK] tap-major
                    const float* __restrict__ dArr,          // [B][NC]
                    const float* __restrict__ bias,          // [NC]
                    const float* __restrict__ sNext,         // [B][NC] (stage0) or null
                    unsigned short* __restrict__ outBf,      // stage0 output
                    float* __restrict__ outF,                // stage1 output
                    int stage) {
  __shared__ __align__(32) unsigned short aTile[TILE_P][APITCH];  // [m][pos]
  __shared__ __align__(32) unsigned short bTile[NC][BPITCH];      // [oc][kk]
  __shared__ float yTile[TILE_P][NC + 1];
  __shared__ float partial[TILE_P][8];
  __shared__ float scaleSh[TILE_P];

  const int t      = threadIdx.x;
  const int b      = blockIdx.y;
  const int p_base = blockIdx.x * TILE_P;
  const int h      = p_base >> 7;         // W = 128
  const int w0p    = p_base & 127;

  const int wave   = t >> 5;
  const int lane   = t & 31;
  const int m_base = (wave & 1) << 4;     // 0 or 16
  const int n_base = (wave >> 1) << 6;    // 0,64,128,192

  // A-staging mapping: channel-within-chunk cc, pixel segment mseg (8 px)
  const int cc   = t & 63;
  const int mseg = t >> 6;                // 0..3 (constant per wave)
  // natural (contiguous) K column, matching the B tile order
  const int posA = cc;

  v8f acc0 = {}, acc1 = {}, acc2 = {}, acc3 = {};

  const unsigned short* xb = xm + (long long)b * NC * PP;
  const unsigned short* wrow = wbT + (size_t)t * KCK;   // this thread's oc row

  for (int kc = 0; kc < NKC; ++kc) {
    const int ck0 = kc * KC;
    const int k   = ck0 >> 8;             // conv tap (fixed in chunk)
    const int c0  = ck0 & 255;            // channel base of chunk
    const int dh  = k / 3 - 1;
    const int dw  = k - (k / 3) * 3 - 1;
    const int h2  = h + dh;
    __syncthreads();

    // --- stage B: thread t = oc, 4x contiguous 16B global -> 16B LDS
    {
      const uint4* src = (const uint4*)(wrow + ck0);
      uint4* dst = (uint4*)&bTile[t][0];
      if (kc + 1 < NKC) __builtin_prefetch(wrow + ck0 + KC, 0, 3);
#pragma unroll
      for (int i = 0; i < 4; ++i) dst[i] = src[i];
    }
    // --- stage A: channel c0+cc, pixels mseg*8..+7 (contiguous run + halo)
    {
      const int c   = c0 + cc;
      const int w2b = w0p + (mseg << 3) + dw;
      unsigned short va[8] __attribute__((aligned(16)));
      if ((unsigned)h2 < HH) {
        const unsigned short* row = xb + (long long)c * PP + (h2 << 7);
        if (dw == 0) {                      // 16B aligned vector load
          *(uint4*)va = *(const uint4*)(row + w2b);
        } else {                            // odd half-offset: coalesced u16
#pragma unroll
          for (int j = 0; j < 8; ++j) {
            int w2 = w2b + j;
            va[j] = ((unsigned)w2 < WW) ? row[w2] : (unsigned short)0;
          }
        }
      } else {
#pragma unroll
        for (int j = 0; j < 8; ++j) va[j] = 0;
      }
#pragma unroll
      for (int j = 0; j < 8; ++j) aTile[(mseg << 3) + j][posA] = va[j];
    }
    __syncthreads();

    // --- 2 K-subchunks of 32: A frag + 4 B frags + 4 WMMAs each
    const int nl   = lane & 15;
    const int hsel = (lane >> 4) << 4;
#pragma unroll
    for (int ko = 0; ko < 2; ++ko) {
      const int off = (ko << 5) + hsel;
      v16u  au = *(const v16u_a16*)&aTile[m_base + nl][off];
      v16bf af = __builtin_bit_cast(v16bf, au);
      v16u bu0 = *(const v16u_a16*)&bTile[n_base + 0  + nl][off];
      v16u bu1 = *(const v16u_a16*)&bTile[n_base + 16 + nl][off];
      v16u bu2 = *(const v16u_a16*)&bTile[n_base + 32 + nl][off];
      v16u bu3 = *(const v16u_a16*)&bTile[n_base + 48 + nl][off];
      acc0 = __builtin_amdgcn_wmma_f32_16x16x32_bf16(false, af, false,
                __builtin_bit_cast(v16bf, bu0), (short)0, acc0, false, false);
      acc1 = __builtin_amdgcn_wmma_f32_16x16x32_bf16(false, af, false,
                __builtin_bit_cast(v16bf, bu1), (short)0, acc1, false, false);
      acc2 = __builtin_amdgcn_wmma_f32_16x16x32_bf16(false, af, false,
                __builtin_bit_cast(v16bf, bu2), (short)0, acc2, false, false);
      acc3 = __builtin_amdgcn_wmma_f32_16x16x32_bf16(false, af, false,
                __builtin_bit_cast(v16bf, bu3), (short)0, acc3, false, false);
    }
  }
  __syncthreads();

  // --- epilogue: demod + bias + leaky*sqrt2 into LDS (C layout: VGPR r ->
  //     M = r + 8*(lane/16), N = lane%16)
  {
    const int nl = lane & 15;
    const int mo = m_base + ((lane >> 4) << 3);
#pragma unroll
    for (int j = 0; j < 4; ++j) {
      int n = n_base + (j << 4) + nl;
      float dv = dArr[b * NC + n];
      float bv = bias[n];
      v8f a = (j == 0) ? acc0 : (j == 1) ? acc1 : (j == 2) ? acc2 : acc3;
#pragma unroll
      for (int r8 = 0; r8 < 8; ++r8) {
        float y = a[r8] * dv + bv;
        y = (y > 0.f ? y : 0.2f * y) * 1.4142135623730951f;
        yTile[mo + r8][n] = y;
      }
    }
  }
  __syncthreads();

  // --- pixel norm across the 256 channels of each spatial m
  {
    int m = t >> 3, seg = t & 7;
    float s = 0.f;
#pragma unroll
    for (int i = 0; i < 32; ++i) { float v = yTile[m][seg * 32 + i]; s += v * v; }
    partial[m][seg] = s;
  }
  __syncthreads();
  if (t < TILE_P) {
    float s = 0.f;
#pragma unroll
    for (int i = 0; i < 8; ++i) s += partial[t][i];
    scaleSh[t] = rsqrtf(s * (1.f / NC) + 1e-8f);
  }
  __syncthreads();

  // --- write out: stage0 -> fused *s1 + bf16 pack; stage1 -> f32 result
  {
    int m  = t & 31;
    int ng = t >> 5;
#pragma unroll
    for (int it = 0; it < 32; ++it) {
      int n = (it << 3) + ng;
      float v = yTile[m][n] * scaleSh[m];
      long long oidx = ((long long)b * NC + n) * PP + p_base + m;
      if (stage == 0) outBf[oidx] = f2bf(v * sNext[b * NC + n]);
      else            outF[oidx]  = v;
    }
  }
}

// ---------------------------------------------------------------------------
extern "C" void kernel_launch(void* const* d_in, const int* in_sizes, int n_in,
                              void* d_out, int out_size, void* d_ws, size_t ws_size,
                              hipStream_t stream) {
  (void)in_sizes; (void)n_in; (void)out_size; (void)ws_size;
  const float* x      = (const float*)d_in[0];
  const float* latent = (const float*)d_in[1];
  const float* w0     = (const float*)d_in[2];
  const float* b0     = (const float*)d_in[3];
  const float* el0w   = (const float*)d_in[4];
  const float* el0b   = (const float*)d_in[5];
  const float* w1     = (const float*)d_in[6];
  const float* b1     = (const float*)d_in[7];
  const float* el1w   = (const float*)d_in[8];
  const float* el1b   = (const float*)d_in[9];

  char* ws = (char*)d_ws;
  float* sArr = (float*)ws;                                   // [2][8][256] f32
  float* dArr = (float*)(ws + 16384);                         // [2][8][256] f32
  unsigned short* wb0 = (unsigned short*)(ws + 32768);        // [256][2304] bf16
  unsigned short* wb1 = wb0 + (size_t)NC * KCK;
  unsigned short* xm0 = (unsigned short*)(ws + 32768 + 2 * (size_t)NC * KCK * 2);
  unsigned short* xm1 = xm0 + (size_t)BATCH * NC * PP;        // each 64 MiB bf16

  style_kernel<<<16, 256, 0, stream>>>(latent, el0w, el0b, el1w, el1b, w0, w1,
                                       sArr, dArr);
  wprep_kernel<<<dim3(256, 2), 256, 0, stream>>>(w0, w1, wb0, wb1);
  modulate_kernel<<<4096, 256, 0, stream>>>(x, sArr, xm0);

  modconv_kernel<<<dim3(PP / TILE_P, BATCH), 256, 0, stream>>>(
      xm0, wb0, dArr, b0, sArr + BATCH * NC, xm1, nullptr, 0);
  modconv_kernel<<<dim3(PP / TILE_P, BATCH), 256, 0, stream>>>(
      xm1, wb1, dArr + BATCH * NC, b1, nullptr, nullptr, (float*)d_out, 1);
}
